// MultiHeadAttentionQuantum_65481071401193
// MI455X (gfx1250) — matrix operand order repaired and checked by reading, not compiled, not run on hardware
//
#include <hip/hip_runtime.h>

typedef __bf16 bf16;
typedef __attribute__((ext_vector_type(16))) __bf16 v16bf;
typedef __attribute__((ext_vector_type(8)))  __bf16 v8bf;
typedef __attribute__((ext_vector_type(8)))  float  v8f;
typedef __attribute__((ext_vector_type(4)))  int    v4i;

#define SEQQ  4096
#define EMB   512
#define NBAT  4
#define KSTR  1040   // LDS K-panel row stride (1024 + 16B pad -> conflict-free b128)
#define VSTR  80     // LDS V-panel col stride (64 + 16B pad -> 20-bank stride)
#define KBYT  (32 * KSTR)    // 33280
#define VBYT  (512 * VSTR)   // 40960

__device__ __forceinline__ v8f wmma_bf16(v16bf a, v16bf b, v8f c) {
  // D = A(16x32 bf16) x B(32x16 bf16) + C(16x16 f32)
  return __builtin_amdgcn_wmma_f32_16x16x32_bf16(false, a, false, b, (short)0, c,
                                                 false, false);
}

#if __has_builtin(__builtin_amdgcn_ds_swizzle)
__device__ __forceinline__ float xor16f(float x) {  // lane L <-> L^16 (SWAPX16)
  return __builtin_bit_cast(float,
      __builtin_amdgcn_ds_swizzle(__builtin_bit_cast(int, x), 0x401F));
}
#else
__device__ __forceinline__ float xor16f(float x) { return __shfl_xor(x, 16, 32); }
#endif

#if __has_builtin(__builtin_amdgcn_ds_bpermute)
__device__ __forceinline__ float bcastf(int srcLane, float x) {
  return __builtin_bit_cast(float,
      __builtin_amdgcn_ds_bpermute(srcLane << 2, __builtin_bit_cast(int, x)));
}
#else
__device__ __forceinline__ float bcastf(int srcLane, float x) {
  return __shfl(x, srcLane, 32);
}
#endif

// ------------------------- async global -> LDS (16B per lane), with fallback
#if __has_builtin(__builtin_amdgcn_global_load_async_to_lds_b128)
#define HAS_ASYNC 1
#else
#define HAS_ASYNC 0
#endif

__device__ __forceinline__ void async_cp16(char* lds, const char* g) {
#if HAS_ASYNC
  __builtin_amdgcn_global_load_async_to_lds_b128(
      (__attribute__((address_space(1))) v4i*)g,
      (__attribute__((address_space(3))) v4i*)lds, 0, 0);
#else
  *(v8bf*)lds = *(const v8bf*)g;   // sync fallback through same LDS path
#endif
}

template <int N>
__device__ __forceinline__ void wait_async() {
#if HAS_ASYNC
#if __has_builtin(__builtin_amdgcn_s_wait_asynccnt)
  __builtin_amdgcn_s_wait_asynccnt(N);
#else
  asm volatile("s_wait_asynccnt %0" ::"i"(N));
#endif
#endif
}

// ---------------------------------------------------------------- W -> bf16
__global__ void wcvt_kernel(const float* __restrict__ w, bf16* __restrict__ wb, int n) {
  int i = blockIdx.x * 256 + threadIdx.x;
  if (i < n) wb[i] = (bf16)w[i];
}

// ------------------------------------------------ q = cos(x+theta) @ W^T + b
// Writes q [16384,512] bf16 (row major) and qT in 32-key panels:
//   qT[b][kb/32][col][32]  (each panel = contiguous 32KB -> linear async copy)
__global__ void proj_kernel(const float* __restrict__ x, const float* __restrict__ theta,
                            const bf16* __restrict__ wb, const float* __restrict__ bias,
                            bf16* __restrict__ q, bf16* __restrict__ qT) {
  const int lane = threadIdx.x & 31;
  const int wave = threadIdx.x >> 5;
  const int waveId = blockIdx.x * 8 + wave;
  const int m0 = (waveId >> 3) * 16;   // 1024 row tiles
  const int n0 = (waveId & 7) * 64;    // 8 column strips of 64
  const int half = lane >> 4;
  const int lm = lane & 15;

  v8f acc[4] = {};
  const float* xrow = x + (size_t)(m0 + lm) * EMB;

#pragma unroll
  for (int kt = 0; kt < 16; ++kt) {
    // ---- A fragment: bf16(cos(x + theta))  (16x32 A layout: two 8-elem K chunks)
    const int kb0 = kt * 32 + half * 16;
    const int kb1 = kt * 32 + half * 16 + 8;
    float4 x0 = *(const float4*)(xrow + kb0);
    float4 x1 = *(const float4*)(xrow + kb0 + 4);
    float4 x2 = *(const float4*)(xrow + kb1);
    float4 x3 = *(const float4*)(xrow + kb1 + 4);
    float4 t0 = *(const float4*)(theta + (kb0 & 63));
    float4 t1 = *(const float4*)(theta + ((kb0 + 4) & 63));
    float4 t2 = *(const float4*)(theta + (kb1 & 63));
    float4 t3 = *(const float4*)(theta + ((kb1 + 4) & 63));
    float xv[16] = {x0.x, x0.y, x0.z, x0.w, x1.x, x1.y, x1.z, x1.w,
                    x2.x, x2.y, x2.z, x2.w, x3.x, x3.y, x3.z, x3.w};
    float tv[16] = {t0.x, t0.y, t0.z, t0.w, t1.x, t1.y, t1.z, t1.w,
                    t2.x, t2.y, t2.z, t2.w, t3.x, t3.y, t3.z, t3.w};
    v16bf a;
#pragma unroll
    for (int e = 0; e < 16; ++e) a[e] = (bf16)__cosf(xv[e] + tv[e]);

    // ---- B fragments: rows of W (q = qc @ W^T -> B[k,n] = W[n,k], contiguous)
#pragma unroll
    for (int nt = 0; nt < 4; ++nt) {
      const bf16* wr = wb + (size_t)(n0 + nt * 16 + lm) * EMB + kt * 32 + half * 16;
      v16bf bf = *(const v16bf*)wr;
      acc[nt] = wmma_bf16(a, bf, acc[nt]);
    }
  }

  // ---- epilogue: + bias, store bf16 q and panel-transposed qT
#pragma unroll
  for (int nt = 0; nt < 4; ++nt) {
    const int col = n0 + nt * 16 + lm;
    const float bv = bias[col];
#pragma unroll
    for (int r = 0; r < 8; ++r) {
      const int row = m0 + r + 8 * half;   // D layout: M = r + 8*(lane/16)
      const float v = acc[nt][r] + bv;
      const bf16 h = (bf16)v;
      q[(size_t)row * EMB + col] = h;
      const int b = row >> 12;             // row / 4096
      const int srow = row & 4095;
      qT[(((size_t)b * 128 + (srow >> 5)) * EMB + col) * 32 + (srow & 31)] = h;
    }
  }
}

// ------------------------------------------- fused flash attention, q=k=v=q
// Per WG (256 thr, 8 waves): 64 query rows. wave = (mt: 16-row tile, ch: 256-col half).
// K/V panels double-buffered in LDS via async global->LDS DMA (ASYNCcnt).
__global__ void attn_kernel(const bf16* __restrict__ q, const bf16* __restrict__ qT,
                            float* __restrict__ out) {
  __shared__ __align__(16) char smK[2][KBYT];
  __shared__ __align__(16) char smV[2][VBYT];

  const int tid = threadIdx.x;
  const int lane = tid & 31;
  const int wave = tid >> 5;
  const int lm = lane & 15;
  const int half = lane >> 4;
  const int mt = wave & 3;
  const int ch = wave >> 2;
  const int bid = blockIdx.x;
  const int b = bid >> 6;                       // 64 blocks per batch
  const int q0 = (bid & 63) * 64 + mt * 16;     // query tile start (within batch)

  const bf16* qb  = q  + (size_t)b * SEQQ * EMB;
  const bf16* qTb = qT + (size_t)b * 128 * EMB * 32;

  // --- stage one 32-key block: K panel (32x512, contiguous rows) and V panel
  //     (contiguous qT panel [512 cols][32 keys]) -> 16 async b128 per thread.
  auto stage = [&](int kb, char* kbuf, char* vbuf) {
    const char* kg = (const char*)(qb + (size_t)kb * EMB);
    const char* vg = (const char*)(qTb + (size_t)(kb >> 5) * EMB * 32);
#pragma unroll
    for (int c = 0; c < 8; ++c) {
      const int j = c * 256 + tid;                       // 16B chunk index, 0..2047
      async_cp16(kbuf + (j >> 6) * KSTR + ((j & 63) << 4), kg + (size_t)j * 16);
      async_cp16(vbuf + (j >> 2) * VSTR + ((j & 3) << 4), vg + (size_t)j * 16);
    }
  };

  // Persistent q B-fragments for sT = K·Qᵀ (lane = query lm, elems = embed)
  v16bf qB[16];
  {
    const bf16* qrow = qb + (size_t)(q0 + lm) * EMB;
#pragma unroll
    for (int kt = 0; kt < 16; ++kt)
      qB[kt] = *(const v16bf*)(qrow + kt * 32 + half * 16);
  }

  v8f o[16] = {};                 // 16 rows x 256 cols f32 accumulator (this half)
  float m_run = -__builtin_inff();
  float l_run = 0.f;
  const int c0 = ch * 256;
  const int mb = half * 8;        // O-row base (D layout M = r + 8*half)
  const float SCALE = 0.125f;     // 1/sqrt(d_k=64)
  const int NBLK = SEQQ / 32;     // 128

  stage(0, smK[0], smV[0]);
  stage(32, smK[1], smV[1]);

  for (int i = 0; i < NBLK; ++i) {
    if (i + 1 < NBLK) wait_async<16>(); else wait_async<0>();
    __syncthreads();

    const char* kc = smK[i & 1];
    const char* vc = smV[i & 1];

    // ---- sT = K_block · Q_tileᵀ : two 16x16 D tiles (key subtiles jt=0,1)
    v8f s0 = {}, s1 = {};
    const char* kr0 = kc + (size_t)lm * KSTR;
    const char* kr1 = kc + (size_t)(16 + lm) * KSTR;
#pragma unroll
    for (int kt = 0; kt < 16; ++kt) {
      const int o0 = kt * 64 + half * 32;        // bytes: embed kt*32 + half*16
      v8bf a0lo = *(const v8bf*)(kr0 + o0);
      v8bf a0hi = *(const v8bf*)(kr0 + o0 + 16);
      v8bf a1lo = *(const v8bf*)(kr1 + o0);
      v8bf a1hi = *(const v8bf*)(kr1 + o0 + 16);
      v16bf a0 = __builtin_shufflevector(a0lo, a0hi, 0, 1, 2, 3, 4, 5, 6, 7,
                                         8, 9, 10, 11, 12, 13, 14, 15);
      v16bf a1 = __builtin_shufflevector(a1lo, a1hi, 0, 1, 2, 3, 4, 5, 6, 7,
                                         8, 9, 10, 11, 12, 13, 14, 15);
      s0 = wmma_bf16(a0, qB[kt], s0);
      s1 = wmma_bf16(a1, qB[kt], s1);
    }

    // ---- online softmax: lane lm owns query lm; 16 keys in-lane + 16 in lane^16
    float p0[8], p1[8];
    float mx = -__builtin_inff();
#pragma unroll
    for (int r = 0; r < 8; ++r) {
      p0[r] = s0[r] * SCALE;
      p1[r] = s1[r] * SCALE;
      mx = fmaxf(mx, fmaxf(p0[r], p1[r]));
    }
    mx = fmaxf(mx, xor16f(mx));
    const float m_new = fmaxf(m_run, mx);
    const float corr = __expf(m_run - m_new);

    v16bf pA;                     // P maps element-for-element onto 16x32 A layout
    float ps = 0.f;
#pragma unroll
    for (int r = 0; r < 8; ++r) {
      const float e0 = __expf(p0[r] - m_new);
      const float e1 = __expf(p1[r] - m_new);
      ps += e0 + e1;
      pA[r] = (bf16)e0;
      pA[r + 8] = (bf16)e1;
    }
    ps += xor16f(ps);
    l_run = l_run * corr + ps;
    m_run = m_new;

    float cb[8];
#pragma unroll
    for (int r = 0; r < 8; ++r) cb[r] = bcastf(mb + r, corr);

    // ---- O = O*corr + P·V ; V fragment: keys 8h..8h+7 and 16+8h..16+8h+7 from LDS panel
#pragma unroll
    for (int nt = 0; nt < 16; ++nt) {
#pragma unroll
      for (int r = 0; r < 8; ++r) o[nt][r] *= cb[r];
      const char* vp = vc + (size_t)(c0 + nt * 16 + lm) * VSTR + half * 16;
      v8bf vlo = *(const v8bf*)vp;
      v8bf vhi = *(const v8bf*)(vp + 32);
      v16bf vB = __builtin_shufflevector(vlo, vhi, 0, 1, 2, 3, 4, 5, 6, 7,
                                         8, 9, 10, 11, 12, 13, 14, 15);
      o[nt] = wmma_bf16(pA, vB, o[nt]);
    }

    __syncthreads();              // everyone done reading buf[i&1]
    if (i + 2 < NBLK) stage((i + 2) * 32, smK[i & 1], smV[i & 1]);
  }

  // ---- finalize: divide rows by l, store f32
  const float inv_l = 1.0f / l_run;
  float lb[8];
#pragma unroll
  for (int r = 0; r < 8; ++r) lb[r] = bcastf(mb + r, inv_l);

#pragma unroll
  for (int nt = 0; nt < 16; ++nt) {
    const int col = c0 + nt * 16 + lm;
#pragma unroll
    for (int r = 0; r < 8; ++r) {
      const int row = q0 + r + mb;
      out[((size_t)b * SEQQ + row) * EMB + col] = o[nt][r] * lb[r];
    }
  }
}

// ---------------------------------------------------------------------------
extern "C" void kernel_launch(void* const* d_in, const int* in_sizes, int n_in,
                              void* d_out, int out_size, void* d_ws, size_t ws_size,
                              hipStream_t stream) {
  const float* x     = (const float*)d_in[0];   // [4,4096,512]
  const float* theta = (const float*)d_in[1];   // [64]
  const float* W     = (const float*)d_in[2];   // [512,512]
  const float* bias  = (const float*)d_in[3];   // [512]
  float* out = (float*)d_out;

  char* ws = (char*)d_ws;
  bf16* qbf = (bf16*)ws;                                   // 16 MiB
  bf16* qT  = (bf16*)(ws + (size_t)16 * 1024 * 1024);      // 16 MiB (32-key panels)
  bf16* wb  = (bf16*)(ws + (size_t)32 * 1024 * 1024);      // 512 KiB

  wcvt_kernel<<<1024, 256, 0, stream>>>(W, wb, EMB * EMB);
  proj_kernel<<<1024, 256, 0, stream>>>(x, theta, wb, bias, qbf, qT);
  attn_kernel<<<256, 256, 0, stream>>>(qbf, qT, out);
}
